// FixedKAN_14585708937626
// MI455X (gfx1250) — compile-verified, matrix operand order checked by reading, not run on hardware
//
#include <hip/hip_runtime.h>

typedef __attribute__((ext_vector_type(16))) _Float16 v16h;
typedef __attribute__((ext_vector_type(8)))  _Float16 v8h;
typedef __attribute__((ext_vector_type(8)))  float    v8f;

#define BATCH    32768
#define INDIM    256
#define DP1      8
#define NF       (INDIM * DP1)   // 2048 features
#define NEURONS  128

#define WAVES          2
#define ROWS_PER_WAVE  16
#define ROWS_PER_BLOCK (WAVES * ROWS_PER_WAVE)  // 32
#define LDS_STRIDE     260   // dwords: 1040B row stride -> 16B aligned, bank-conflict-free

// ---------------- Kernel 1: c_sum[f] = sum_n coeffs[n][f], store as f16 ----------------
__global__ __launch_bounds__(256) void kan_reduce_coeffs(const float* __restrict__ coeffs,
                                                         _Float16* __restrict__ csum) {
  int f = blockIdx.x * blockDim.x + threadIdx.x;
  if (f >= NF) return;
  float s = 0.f;
#pragma unroll 8
  for (int n = 0; n < NEURONS; ++n) s += coeffs[(size_t)n * NF + f];
  csum[f] = (_Float16)s;
}

// ---------------- Kernel 2: out[b] = sum_f T(x)[b,f] * c_sum[f] via WMMA ----------------
__global__ __launch_bounds__(WAVES * 32) void kan_main(const float* __restrict__ x,
                                                       const _Float16* __restrict__ csum,
                                                       float* __restrict__ out) {
  __shared__ float    tile[ROWS_PER_BLOCK * LDS_STRIDE];            // ~33 KB
  __shared__ __attribute__((aligned(16))) _Float16 cs[NF];          // 4 KB

  const int t        = threadIdx.x;            // 0..63
  const int row_base = blockIdx.x * ROWS_PER_BLOCK;

  // Stage c_sum (1024 dwords) cooperatively.
  {
    const unsigned* src = (const unsigned*)csum;
    unsigned*       dst = (unsigned*)cs;
#pragma unroll
    for (int j = 0; j < (NF / 2) / (WAVES * 32); ++j)
      dst[j * (WAVES * 32) + t] = src[j * (WAVES * 32) + t];
  }

  // Async-copy x tile (32 rows x 256 f32) global -> LDS, 16B per lane per issue.
  // Iteration j copies row j: 64 lanes cover 64 x b128 = 1KB fully coalesced.
#pragma unroll 4
  for (int j = 0; j < ROWS_PER_BLOCK; ++j) {
    const int    c16 = t;  // 16B chunk within row
    const float* g   = x + (size_t)(row_base + j) * INDIM + c16 * 4;
    unsigned lds_off =
        (unsigned)(unsigned long long)(unsigned long)(uintptr_t)&tile[j * LDS_STRIDE + c16 * 4];
    asm volatile("global_load_async_to_lds_b128 %0, %1, off"
                 :
                 : "v"(lds_off), "v"(g)
                 : "memory");
  }
  asm volatile("s_wait_asynccnt 0" ::: "memory");
  __syncthreads();

  const int lane = t & 31;
  const int wv   = t >> 5;
  const int g    = lane >> 4;   // lane group: selects K offset per ISA 16-bit layouts
  const int m    = lane & 15;   // A-matrix row within tile
  const float* xrow = &tile[(wv * ROWS_PER_WAVE + m) * LDS_STRIDE];

  v8f acc = {0.f, 0.f, 0.f, 0.f, 0.f, 0.f, 0.f, 0.f};

  // 64 chunks of K=32 (4 input dims x 8 Chebyshev degrees, dim-major like reference)
#pragma unroll 4
  for (int c = 0; c < INDIM / 4; ++c) {
    // This lane supplies dims (4c+g) [halves 0..7] and (4c+2+g) [halves 8..15]:
    //   g=0: K 0..7 & 16..23 ; g=1: K 8..15 & 24..31  (ISA 16-bit A 16x32 layout)
    const float xa  = xrow[4 * c + g];
    const float xb  = xrow[4 * c + 2 + g];
    const float x2a = xa + xa;
    const float x2b = xb + xb;

    float Ta[DP1], Tb[DP1];
    Ta[0] = 1.f; Ta[1] = xa;
    Tb[0] = 1.f; Tb[1] = xb;
#pragma unroll
    for (int d = 2; d < DP1; ++d) {
      Ta[d] = __builtin_fmaf(x2a, Ta[d - 1], -Ta[d - 2]);
      Tb[d] = __builtin_fmaf(x2b, Tb[d - 1], -Tb[d - 2]);
    }

    v16h a;
#pragma unroll
    for (int h = 0; h < 8; ++h) {
      a[h]     = (_Float16)Ta[h];
      a[h + 8] = (_Float16)Tb[h];
    }

    // B: ALL lanes broadcast-load the same c_sum chunk for their lane group's K set.
    // Then B[k,n] = c[k] for every column n, so every D column holds sum_k A[m,k]c[k];
    // we read column 0. No masking -> no exec-predicated micro-loads, and the result
    // is independent of the B-operand lane<->N mapping. Two 16B broadcast LDS loads.
    const v8h* cp  = (const v8h*)&cs[32 * c + 8 * g];  // 16B aligned
    const v8h  blo = cp[0];   // K = 8g + 0..7
    const v8h  bhi = cp[2];   // K = 8g + 16..23
    const v16h b   = __builtin_shufflevector(blo, bhi,
                                             0, 1, 2, 3, 4, 5, 6, 7,
                                             8, 9, 10, 11, 12, 13, 14, 15);

    acc = __builtin_amdgcn_wmma_f32_16x16x32_f16(
        /*neg_a=*/false, a, /*neg_b=*/false, b,
        /*c_mod=*/(short)0, acc, /*reuse_a=*/false, /*reuse_b=*/false);
  }

  // D column 0: lane 0 holds rows 0..7 (VGPR j -> M=j), lane 16 holds rows 8..15.
  if (m == 0) {
    const int r0 = row_base + wv * ROWS_PER_WAVE + g * 8;
#pragma unroll
    for (int j = 0; j < 8; ++j) out[r0 + j] = acc[j];
  }
}

extern "C" void kernel_launch(void* const* d_in, const int* in_sizes, int n_in,
                              void* d_out, int out_size, void* d_ws, size_t ws_size,
                              hipStream_t stream) {
  const float* x      = (const float*)d_in[0];  // [32768, 256]
  const float* coeffs = (const float*)d_in[1];  // [128, 2048]
  _Float16*    csum   = (_Float16*)d_ws;        // 4 KB scratch
  float*       out    = (float*)d_out;          // [32768, 1]

  kan_reduce_coeffs<<<NF / 256, 256, 0, stream>>>(coeffs, csum);
  kan_main<<<BATCH / ROWS_PER_BLOCK, WAVES * 32, 0, stream>>>(x, csum, out);
}
